// SelfAttentionBlock_GPT_65283502899741
// MI455X (gfx1250) — compile-verified, matrix operand order checked by reading, not run on hardware
//
#include <hip/hip_runtime.h>

// ---------------------------------------------------------------------------
// Self-attention block (B=8, C=256, H=W=64, QK=32) for gfx1250 (MI455X).
// Fused flash-attention with v_wmma_f32_16x16x32_bf16, f32 accumulate.
// K/V chunks staged in LDS via the Tensor Data Mover (tensor_load_to_lds),
// double-buffered, shared by 8 waves (64 queries) per block.
// ---------------------------------------------------------------------------

typedef __attribute__((ext_vector_type(16))) __bf16 v16bf;
typedef __attribute__((ext_vector_type(8)))  __bf16 v8bf;
typedef __attribute__((ext_vector_type(8)))  float  v8f;
typedef __attribute__((ext_vector_type(4)))  unsigned int v4u;
typedef __attribute__((ext_vector_type(8)))  int v8i;
typedef __attribute__((ext_vector_type(4)))  int v4i;

#define BATCH 8
#define CHN   256
#define NPIX  4096
#define QKD   32

__device__ __forceinline__ v8f wmma_bf16(v16bf a, v16bf b, v8f c) {
  // D = A(16x32 bf16) * B(32x16 bf16) + C(16x16 f32)
  return __builtin_amdgcn_wmma_f32_16x16x32_bf16(
      /*neg_a=*/false, a, /*neg_b=*/false, b,
      /*c_mod=*/(short)0, c, /*reuse_a=*/false, /*reuse_b=*/false);
}

// 16-bit A/B matrix VGPR layout: per lane, K groups [base..base+7] and
// [base+16..base+23] are contiguous (base = 8 for lanes 16..31).
// p must already include the +base element offset.
__device__ __forceinline__ v16bf ld_pair(const __bf16* p) {
  union { v16bf v; v8bf h[2]; } u;
  u.h[0] = *(const v8bf*)(p);
  u.h[1] = *(const v8bf*)(p + 16);
  return u.v;
}

// ---------------------------------------------------------------------------
// TDM: 2-D tile load Global->LDS via Tensor DMA Descriptor (ISA 8.3-8.5).
// data_size = 2 bytes (bf16). Dims/strides in elements, addresses in bytes.
// Group0: [1:0]count=1 | [63:32]lds_addr | [120:64]global_addr | [127:126]type=2
// Group1: [17:16]data_size=1 | [79:48]tensor_dim0 | [111:80]tensor_dim1 |
//         [127:112]tile_dim0 | [143:128]tile_dim1 | [207:160]dim0_stride
// Groups 2/3 zero (<=2-D tile).
// ---------------------------------------------------------------------------
__device__ __forceinline__ void tdm_load_2d(unsigned lds_off, const void* gptr,
                                            unsigned tensor_d0, unsigned tensor_d1,
                                            unsigned tile_d0, unsigned tile_d1,
                                            unsigned long long stride0) {
  unsigned long long ga = (unsigned long long)(size_t)gptr;
  v4u g0;
  g0[0] = 1u;                                            // count=1, user D#
  g0[1] = lds_off;                                       // lds_addr (bytes)
  g0[2] = (unsigned)(ga & 0xFFFFFFFFu);                  // global_addr[31:0]
  g0[3] = (unsigned)((ga >> 32) & 0x01FFFFFFu) | (2u << 30); // addr[56:32]|type=2
  v8i g1;
  g1[0] = (int)(1u << 16);                               // data_size=1 -> 2B
  g1[1] = (int)((tensor_d0 & 0xFFFFu) << 16);            // dim0[15:0] @ [63:48]
  g1[2] = (int)((tensor_d0 >> 16) | ((tensor_d1 & 0xFFFFu) << 16));
  g1[3] = (int)((tensor_d1 >> 16) | (tile_d0 << 16));    // dim1 hi | tile_dim0
  g1[4] = (int)(tile_d1 & 0xFFFFu);                      // tile_dim1; tile_dim2=0
  g1[5] = (int)(stride0 & 0xFFFFFFFFull);                // dim0_stride lo32
  g1[6] = (int)((stride0 >> 32) & 0xFFFFull);            // dim0_stride hi16
  g1[7] = 0;
  v4i z4 = {0, 0, 0, 0};
#if defined(__clang_major__) && (__clang_major__ >= 23)
  v8i z8 = {0, 0, 0, 0, 0, 0, 0, 0};
  __builtin_amdgcn_tensor_load_to_lds(g0, g1, z4, z4, z8, 0);
#else
  __builtin_amdgcn_tensor_load_to_lds(g0, g1, z4, z4, 0);
#endif
}

// ---------------------------------------------------------------------------
// Kernel 1: x (B,C,N) f32 -> xT (B,N,C) bf16
// ---------------------------------------------------------------------------
__global__ __launch_bounds__(256) void prep_x_kernel(
    const float* __restrict__ x, __bf16* __restrict__ xT) {
  size_t tid = (size_t)blockIdx.x * 256 + threadIdx.x;
  size_t total = (size_t)BATCH * CHN * NPIX;
  if (tid >= total) return;
  int n = (int)(tid % NPIX);
  size_t bc = tid / NPIX;
  int c = (int)(bc % CHN);
  int b = (int)(bc / CHN);
  xT[((size_t)b * NPIX + n) * CHN + c] = (__bf16)x[tid];
}

// ---------------------------------------------------------------------------
// Kernel 2: concat Wq(32x256), Wk(32x256), Wv(256x256) -> Wcat (320x256) bf16
// ---------------------------------------------------------------------------
__global__ __launch_bounds__(256) void prep_w_kernel(
    const float* __restrict__ Wq, const float* __restrict__ Wk,
    const float* __restrict__ Wv, __bf16* __restrict__ Wcat) {
  int tid = blockIdx.x * 256 + threadIdx.x;   // 320*256 = 81920 total
  if (tid < 8192)        Wcat[tid] = (__bf16)Wq[tid];
  else if (tid < 16384)  Wcat[tid] = (__bf16)Wk[tid - 8192];
  else if (tid < 81920)  Wcat[tid] = (__bf16)Wv[tid - 16384];
}

// ---------------------------------------------------------------------------
// Kernel 3: projections via WMMA.
// Per wave: one 16-pixel x 16-output tile, K=256 in 8 chunks of 32.
// Outputs: Q (B,N,32) bf16, Kt (B,N,32) bf16, V (B,C,N) bf16.
// ---------------------------------------------------------------------------
__global__ __launch_bounds__(128) void proj_kernel(
    const __bf16* __restrict__ xT, const __bf16* __restrict__ Wcat,
    __bf16* __restrict__ Q, __bf16* __restrict__ Kt, __bf16* __restrict__ V) {
  int wave  = blockIdx.x * 4 + (threadIdx.x >> 5);
  int lane  = threadIdx.x & 31;
  int l15   = lane & 15;
  int hf    = lane >> 4;                   // 0 for lanes 0-15, 1 for 16-31
  int otile = wave % 20;                   // 0-1 Q, 2-3 Kt, 4-19 V
  int ptile = (wave / 20) & 255;
  int b     = wave / (20 * 256);

  const __bf16* arow = xT   + ((size_t)b * NPIX + ptile * 16 + l15) * CHN + hf * 8;
  const __bf16* brow = Wcat + ((size_t)(otile * 16 + l15)) * CHN + hf * 8;

  v8f d;
  #pragma unroll
  for (int r = 0; r < 8; ++r) d[r] = 0.0f;

  #pragma unroll
  for (int cc = 0; cc < CHN; cc += 32) {
    v16bf a = ld_pair(arow + cc);
    v16bf w = ld_pair(brow + cc);
    d = wmma_bf16(a, w, d);
  }

  // C/D layout: VGPR r holds row (r + 8*hf), column l15.
  #pragma unroll
  for (int r = 0; r < 8; ++r) {
    int pix = ptile * 16 + r + hf * 8;
    __bf16 val = (__bf16)d[r];
    if (otile < 2) {
      Q[((size_t)b * NPIX + pix) * QKD + otile * 16 + l15] = val;
    } else if (otile < 4) {
      Kt[((size_t)b * NPIX + pix) * QKD + (otile - 2) * 16 + l15] = val;
    } else {
      V[((size_t)b * CHN + (otile - 4) * 16 + l15) * NPIX + pix] = val;
    }
  }
}

// ---------------------------------------------------------------------------
// Kernel 4: fused flash attention + epilogue (gamma*out + x).
// Block = 256 threads = 8 waves, covers 64 queries x all 256 channels:
//   wave wid: query tile qt = qg*4 + (wid>>1), channel half = 128*(wid&1).
// Per key-chunk of 32: TDM stages K (32x32) and V (256x32) tiles into LDS
// (double-buffered, issued by wave 0, overlapped with compute).
// ---------------------------------------------------------------------------
__global__ __launch_bounds__(256) void flash_kernel(
    const __bf16* __restrict__ Q, const __bf16* __restrict__ Kt,
    const __bf16* __restrict__ V, const float* __restrict__ x,
    const float* __restrict__ gamma, float* __restrict__ out) {
  __shared__ __bf16 Vlds[2][CHN * 32];   // [buf][ch*32 + key]   16KB each
  __shared__ __bf16 Klds[2][32 * 32];    // [buf][key*32 + dim]   2KB each
  __shared__ __bf16 Pbuf[8][16 * 32];    // per-wave P staging    8KB

  int wid  = threadIdx.x >> 5;
  int lane = threadIdx.x & 31;
  int l15  = lane & 15;
  int hf   = lane >> 4;
  int b    = blockIdx.x >> 6;
  int qg   = blockIdx.x & 63;
  int qt   = qg * 4 + (wid >> 1);
  int ch0  = (wid & 1) * 128;
  const float scale = 0.17677669529663687f;  // 1/sqrt(32)

  const __bf16* Vbase  = V  + (size_t)b * CHN * NPIX;
  const __bf16* Ktbase = Kt + (size_t)b * NPIX * QKD;

  // Q tile in A layout (16 queries x 32 dims)
  v16bf qA = ld_pair(Q + ((size_t)b * NPIX + qt * 16 + l15) * QKD + hf * 8);

  v8f acc[8];
  float rowm[8], rowl[8];
  #pragma unroll
  for (int t = 0; t < 8; ++t)
    #pragma unroll
    for (int r = 0; r < 8; ++r) acc[t][r] = 0.0f;
  #pragma unroll
  for (int r = 0; r < 8; ++r) { rowm[r] = -1e30f; rowl[r] = 0.0f; }

  __bf16* myP = &Pbuf[wid][0];

  // Prologue: stage chunk 0 via TDM.
  if (wid == 0) {
    tdm_load_2d((unsigned)(size_t)(void*)&Vlds[0][0], Vbase,
                NPIX, CHN, 32, CHN, NPIX);          // V tile: 256 rows x 32 keys
    tdm_load_2d((unsigned)(size_t)(void*)&Klds[0][0], Ktbase,
                QKD, NPIX, QKD, 32, QKD);           // K tile: 32 rows x 32 dims
    __builtin_amdgcn_s_wait_tensorcnt(0);
  }
  __syncthreads();

  for (int kc = 0; kc < NPIX / 32; ++kc) {
    int cur = kc & 1;
    // Prefetch next chunk into the other buffer while computing this one.
    if ((kc + 1 < NPIX / 32) && (wid == 0)) {
      int nk0 = (kc + 1) * 32;
      tdm_load_2d((unsigned)(size_t)(void*)&Vlds[cur ^ 1][0], Vbase + nk0,
                  NPIX, CHN, 32, CHN, NPIX);
      tdm_load_2d((unsigned)(size_t)(void*)&Klds[cur ^ 1][0],
                  Ktbase + (size_t)nk0 * QKD, QKD, NPIX, QKD, 32, QKD);
    }

    // ---- QK^T: scores for 16 queries x 32 keys ----
    const __bf16* kp = &Klds[cur][0];
    v16bf kB0 = ld_pair(kp + (size_t)(l15) * 32 + hf * 8);
    v16bf kB1 = ld_pair(kp + (size_t)(l15 + 16) * 32 + hf * 8);

    v8f z;
    #pragma unroll
    for (int r = 0; r < 8; ++r) z[r] = 0.0f;
    v8f s0 = wmma_bf16(qA, kB0, z);
    v8f s1 = wmma_bf16(qA, kB1, z);

    // ---- online softmax (stats replicated per 16-lane half, row = r+8*hf) ----
    #pragma unroll
    for (int r = 0; r < 8; ++r) {
      float a0 = s0[r] * scale;
      float a1 = s1[r] * scale;
      float mx = fmaxf(a0, a1);
      #pragma unroll
      for (int off = 8; off >= 1; off >>= 1)
        mx = fmaxf(mx, __shfl_xor(mx, off, 32));
      float newm  = fmaxf(rowm[r], mx);
      float alpha = __expf(rowm[r] - newm);
      float p0 = __expf(a0 - newm);
      float p1 = __expf(a1 - newm);
      float rs = p0 + p1;
      #pragma unroll
      for (int off = 8; off >= 1; off >>= 1)
        rs += __shfl_xor(rs, off, 32);
      rowl[r] = rowl[r] * alpha + rs;
      rowm[r] = newm;
      #pragma unroll
      for (int t = 0; t < 8; ++t) acc[t][r] *= alpha;
      s0[r] = p0;
      s1[r] = p1;
    }

    // ---- C-layout -> A-layout conversion of P via per-wave LDS tile ----
    #pragma unroll
    for (int r = 0; r < 8; ++r) {
      int row = r + hf * 8;
      myP[row * 32 + l15]      = (__bf16)s0[r];
      myP[row * 32 + 16 + l15] = (__bf16)s1[r];
    }
    asm volatile("s_wait_dscnt 0x0" ::: "memory");  // DS in-order per wave
    v16bf pA;
    {
      union { v16bf v; v8bf h[2]; } u;
      u.h[0] = *(const v8bf*)(myP + l15 * 32 + hf * 8);
      u.h[1] = *(const v8bf*)(myP + l15 * 32 + hf * 8 + 16);
      pA = u.v;
    }
    asm volatile("" ::: "memory");

    // ---- PV: 8 channel tiles of 16, V from LDS ----
    const __bf16* vp = &Vlds[cur][0];
    #pragma unroll
    for (int t = 0; t < 8; ++t) {
      int chn = ch0 + t * 16 + l15;
      v16bf vB = ld_pair(vp + (size_t)chn * 32 + hf * 8);
      acc[t] = wmma_bf16(pA, vB, acc[t]);
    }

    // Publish next chunk's LDS tiles to all waves.
    if (wid == 0) __builtin_amdgcn_s_wait_tensorcnt(0);
    __syncthreads();
  }

  // Epilogue: out = gamma * (acc / l) + x
  float g = gamma[0];
  #pragma unroll
  for (int t = 0; t < 8; ++t) {
    int chn = ch0 + t * 16 + l15;
    #pragma unroll
    for (int r = 0; r < 8; ++r) {
      int pix = qt * 16 + r + hf * 8;
      size_t idx = ((size_t)b * CHN + chn) * NPIX + pix;
      out[idx] = g * (acc[t][r] / rowl[r]) + x[idx];
    }
  }
}

// ---------------------------------------------------------------------------
// Host launch
// ---------------------------------------------------------------------------
extern "C" void kernel_launch(void* const* d_in, const int* in_sizes, int n_in,
                              void* d_out, int out_size, void* d_ws, size_t ws_size,
                              hipStream_t stream) {
  (void)in_sizes; (void)n_in; (void)out_size; (void)ws_size;
  const float* x     = (const float*)d_in[0];
  const float* Wq    = (const float*)d_in[1];
  const float* Wk    = (const float*)d_in[2];
  const float* Wv    = (const float*)d_in[3];
  const float* gamma = (const float*)d_in[4];

  char* ws = (char*)d_ws;
  size_t off = 0;
  __bf16* xT   = (__bf16*)(ws + off); off += (size_t)BATCH * NPIX * CHN * 2;  // 16 MB
  __bf16* Wcat = (__bf16*)(ws + off); off += (size_t)320 * 256 * 2;           // 160 KB
  __bf16* Qb   = (__bf16*)(ws + off); off += (size_t)BATCH * NPIX * QKD * 2;  // 2 MB
  __bf16* Ktb  = (__bf16*)(ws + off); off += (size_t)BATCH * NPIX * QKD * 2;  // 2 MB
  __bf16* Vb   = (__bf16*)(ws + off); off += (size_t)BATCH * CHN * NPIX * 2;  // 16 MB

  // 1) transpose/convert x
  prep_x_kernel<<<(BATCH * CHN * NPIX) / 256, 256, 0, stream>>>(x, xT);
  // 2) convert/concat weights
  prep_w_kernel<<<320, 256, 0, stream>>>(Wq, Wk, Wv, Wcat);
  // 3) QKV projections: 8 b * 256 ptiles * 20 otiles = 40960 waves / 4 = 10240 blocks
  proj_kernel<<<10240, 128, 0, stream>>>(xT, Wcat, Qb, Ktb, Vb);
  // 4) fused flash attention: 8 b * 64 query-groups (64 q each) = 512 blocks
  flash_kernel<<<512, 256, 0, stream>>>(Qb, Ktb, Vb, x, gamma, (float*)d_out);
}
